// HierarchicalModel_59588376264967
// MI455X (gfx1250) — matrix operand order repaired and checked by reading, compiled, not run on hardware
//
#include <hip/hip_runtime.h>
#include <math.h>

typedef __attribute__((ext_vector_type(16))) _Float16 v16h;
typedef __attribute__((ext_vector_type(8)))  _Float16 v8h;
typedef __attribute__((ext_vector_type(8)))  float    v8f;

// ---------------------------------------------------------------------------
// WMMA fragment loaders (wave32, 16x16x32 f16)
// A (16x32, MxK) row-major with leading dim ld:
//   lanes 0-15 : row M=lane,     K = 0..7 (v0-3) and 16..23 (v4-7)
//   lanes 16-31: row M=lane-16,  K = 8..15        and 24..31
// ---------------------------------------------------------------------------
__device__ inline v16h frag_a(const _Float16* __restrict__ tile, int ld) {
  int lane = threadIdx.x & 31;
  const _Float16* p = tile + (size_t)(lane & 15) * ld + ((lane >> 4) << 3);
  v8h lo = *(const v8h*)p;
  v8h hi = *(const v8h*)(p + 16);
  v16h f;
#pragma unroll
  for (int i = 0; i < 8; ++i) { f[i] = lo[i]; f[i + 8] = hi[i]; }
  return f;
}

// B (32x16, KxN) supplied as B^T row-major [N][K] with leading dim ld:
//   lanes 0-15 : col N=lane,     K = 0..7 and 16..23
//   lanes 16-31: col N=lane-16,  K = 8..15 and 24..31
__device__ inline v16h frag_b(const _Float16* __restrict__ tileT, int ld) {
  int lane = threadIdx.x & 31;
  const _Float16* p = tileT + (size_t)(lane & 15) * ld + ((lane >> 4) << 3);
  v8h lo = *(const v8h*)p;
  v8h hi = *(const v8h*)(p + 16);
  v16h f;
#pragma unroll
  for (int i = 0; i < 8; ++i) { f[i] = lo[i]; f[i + 8] = hi[i]; }
  return f;
}

// Templated GEMM body: FULL = all 4 N-subtiles valid (branch-free hot path).
template <bool FULL>
__device__ inline void gemm_body(
    const _Float16* __restrict__ A, const _Float16* __restrict__ Bt,
    float* __restrict__ C, int N, int K, int m0, int n0,
    const float* __restrict__ bias, int act, int acc)
{
  v8f c[2][4];
#pragma unroll
  for (int i = 0; i < 2; ++i)
#pragma unroll
    for (int j = 0; j < 4; ++j)
#pragma unroll
      for (int e = 0; e < 8; ++e) c[i][j][e] = 0.0f;

  const _Float16* a0p = A + (size_t)m0 * K;
  const _Float16* a1p = A + (size_t)(m0 + 16) * K;

  for (int k = 0; k < K; k += 32) {
    v16h a0 = frag_a(a0p + k, K);
    v16h a1 = frag_a(a1p + k, K);
#pragma unroll
    for (int j = 0; j < 4; ++j) {
      if (FULL || (n0 + 16 * j < N)) {
        v16h bf = frag_b(Bt + (size_t)(n0 + 16 * j) * K + k, K);
        c[0][j] = __builtin_amdgcn_wmma_f32_16x16x32_f16(
            false, a0, false, bf, (short)0, c[0][j], false, false);
        c[1][j] = __builtin_amdgcn_wmma_f32_16x16x32_f16(
            false, a1, false, bf, (short)0, c[1][j], false, false);
      }
    }
  }

  // C/D layout: VGPR r -> M = r (lanes 0-15) / 8+r (lanes 16-31); N = lane&15
  int lane = threadIdx.x;
  int lc = lane & 15;
  int hi8 = (lane >> 4) << 3;
#pragma unroll
  for (int i = 0; i < 2; ++i) {
#pragma unroll
    for (int j = 0; j < 4; ++j) {
      if (FULL || (n0 + 16 * j < N)) {
        int col = n0 + 16 * j + lc;
#pragma unroll
        for (int r = 0; r < 8; ++r) {
          int row = m0 + 16 * i + hi8 + r;
          float v = c[i][j][r];
          if (bias) v += bias[col];
          if (act)  v = v / (1.0f + __expf(-v));          // SiLU
          float* dst = C + (size_t)row * N + col;
          if (acc) *dst += v; else *dst = v;
        }
      }
    }
  }
}

// ---------------------------------------------------------------------------
// Batched GEMM: C[b] (+)= A[b] (MxK,f16 row-major) * Bt[b]^T (Bt is NxK f16)
// One wave per 32(M) x 64(N) tile. f32 accumulate in WMMA.
// Requires: M % 32 == 0, K % 32 == 0, N % 16 == 0.
// ---------------------------------------------------------------------------
__global__ __launch_bounds__(32) void gemm_f16_wmma(
    const _Float16* __restrict__ A, const _Float16* __restrict__ Bt,
    float* __restrict__ C, int M, int N, int K,
    long long strideA, long long strideB, long long strideC,
    const float* __restrict__ bias, int act, int acc)
{
  int n0 = blockIdx.x * 64;
  int m0 = blockIdx.y * 32;
  long long bz = blockIdx.z;
  A  += bz * strideA;
  Bt += bz * strideB;
  C  += bz * strideC;
  if (n0 + 64 <= N) gemm_body<true >(A, Bt, C, N, K, m0, n0, bias, act, acc);
  else              gemm_body<false>(A, Bt, C, N, K, m0, n0, bias, act, acc);
}

// ---------------------------------------------------------------------------
// RMSNorm over rows of length D; optional f32 and f16 outputs (in-place OK).
// ---------------------------------------------------------------------------
__global__ void rmsnorm_kernel(const float* __restrict__ x, const float* __restrict__ w,
                               float* __restrict__ of32, _Float16* __restrict__ of16, int D)
{
  int row = blockIdx.x;
  const float* xr = x + (size_t)row * D;
  __shared__ float red[256];
  float s = 0.0f;
  for (int i = threadIdx.x; i < D; i += 256) { float v = xr[i]; s += v * v; }
  red[threadIdx.x] = s; __syncthreads();
  for (int off = 128; off > 0; off >>= 1) {
    if ((int)threadIdx.x < off) red[threadIdx.x] += red[threadIdx.x + off];
    __syncthreads();
  }
  float scale = rsqrtf(red[0] / (float)D + 1e-5f);
  for (int i = threadIdx.x; i < D; i += 256) {
    float v = xr[i] * scale * w[i];
    if (of32) of32[(size_t)row * D + i] = v;
    if (of16) of16[(size_t)row * D + i] = (_Float16)v;
  }
}

// f32 [R][C] -> f16 [C][R]  (weights: makes B^T contiguous for frag_b)
__global__ void transpose_f32_to_f16(const float* __restrict__ in,
                                     _Float16* __restrict__ out, int R, int C)
{
  size_t idx = (size_t)blockIdx.x * 256 + threadIdx.x;
  if (idx >= (size_t)R * C) return;
  int r = (int)(idx / C), c = (int)(idx % C);
  out[(size_t)c * R + r] = (_Float16)in[idx];
}

// qkv f32 [B*L][3D] -> rope(q),rope(k) f16 [B,H,L,hd] and v^T f16 [B,H,hd,L]
__global__ void rope_split_kernel(const float* __restrict__ qkv,
                                  const float* __restrict__ cosb, const float* __restrict__ sinb,
                                  _Float16* __restrict__ qh, _Float16* __restrict__ kh,
                                  _Float16* __restrict__ vT, int D, int hd, int H, int L)
{
  int l = blockIdx.x, h = blockIdx.y, b = blockIdx.z;
  int d = threadIdx.x; if (d >= hd) return;
  int half = hd >> 1;
  int i = (d < half) ? d : d - half;
  float cc = cosb[(size_t)l * half + i];
  float ss = sinb[(size_t)l * half + i];
  size_t rowb = (size_t)(b * L + l) * (3 * D);
  float q1 = qkv[rowb + h * hd + i];
  float q2 = qkv[rowb + h * hd + i + half];
  float k1 = qkv[rowb + D + h * hd + i];
  float k2 = qkv[rowb + D + h * hd + i + half];
  float qv = (d < half) ? q1 * cc - q2 * ss : q1 * ss + q2 * cc;
  float kv = (d < half) ? k1 * cc - k2 * ss : k1 * ss + k2 * cc;
  float vv = qkv[rowb + 2 * D + h * hd + d];
  size_t bh = (size_t)(b * H + h);
  qh[(bh * L + l) * hd + d] = (_Float16)qv;
  kh[(bh * L + l) * hd + d] = (_Float16)kv;
  vT[(bh * hd + d) * L + l] = (_Float16)vv;
}

// causal softmax row-wise: P = softmax(S*scale masked), f16 out
__global__ void softmax_causal(const float* __restrict__ S, _Float16* __restrict__ P,
                               int L, float scale)
{
  int q = blockIdx.x, bh = blockIdx.y;
  const float* row = S + ((size_t)bh * L + q) * L;
  _Float16* orow = P + ((size_t)bh * L + q) * L;
  __shared__ float red[256];
  float m = -1e30f;
  for (int i = threadIdx.x; i <= q; i += 256) m = fmaxf(m, row[i] * scale);
  red[threadIdx.x] = m; __syncthreads();
  for (int off = 128; off > 0; off >>= 1) {
    if ((int)threadIdx.x < off) red[threadIdx.x] = fmaxf(red[threadIdx.x], red[threadIdx.x + off]);
    __syncthreads();
  }
  m = red[0]; __syncthreads();
  float s = 0.0f;
  for (int i = threadIdx.x; i <= q; i += 256) s += __expf(row[i] * scale - m);
  red[threadIdx.x] = s; __syncthreads();
  for (int off = 128; off > 0; off >>= 1) {
    if ((int)threadIdx.x < off) red[threadIdx.x] += red[threadIdx.x + off];
    __syncthreads();
  }
  float inv = 1.0f / red[0];
  for (int i = threadIdx.x; i < L; i += 256)
    orow[i] = (_Float16)((i <= q) ? __expf(row[i] * scale - m) * inv : 0.0f);
}

// O f32 [B,H,L,hd] -> f16 [B,L,D]
__global__ void merge_heads(const float* __restrict__ O, _Float16* __restrict__ om,
                            int H, int L, int hd, int D, size_t tot)
{
  size_t idx = (size_t)blockIdx.x * 256 + threadIdx.x;
  if (idx >= tot) return;
  int d = (int)(idx % hd); size_t t = idx / hd;
  int l = (int)(t % L); t /= L;
  int h = (int)(t % H); int b = (int)(t / H);
  om[(size_t)(b * L + l) * D + h * hd + d] = (_Float16)O[idx];
}

__global__ void silu_mul(const float* __restrict__ g, const float* __restrict__ u,
                         _Float16* __restrict__ t, size_t n)
{
  size_t i = (size_t)blockIdx.x * 256 + threadIdx.x;
  if (i >= n) return;
  float gv = g[i];
  t[i] = (_Float16)((gv / (1.0f + __expf(-gv))) * u[i]);
}

// probs[row] = sigmoid(dot(r1[row], w2) + b2)
__global__ void router_logit(const float* __restrict__ r1, const float* __restrict__ w2,
                             const float* __restrict__ b2, float* __restrict__ probs, int D)
{
  int row = blockIdx.x;
  __shared__ float red[256];
  float s = 0.0f;
  for (int i = threadIdx.x; i < D; i += 256) s += r1[(size_t)row * D + i] * w2[i];
  red[threadIdx.x] = s; __syncthreads();
  for (int off = 128; off > 0; off >>= 1) {
    if ((int)threadIdx.x < off) red[threadIdx.x] += red[threadIdx.x + off];
    __syncthreads();
  }
  if (threadIdx.x == 0) {
    float lg = red[0] + b2[0];
    probs[row] = 1.0f / (1.0f + __expf(-lg));
  }
}

// per-batch sequential scan: hard mask (pos0 forced), cumsum -> chunk ids,
// selected positions in order (stable argsort(-hard) over valid entries)
__global__ void router_scan(const float* __restrict__ probs, int* __restrict__ chunk_ids,
                            int* __restrict__ sel_pos, int L)
{
  int b = blockIdx.x;
  if (threadIdx.x != 0) return;
  int cum = 0;
  for (int l = 0; l < L; ++l) {
    int hsel = (l == 0) || (probs[b * L + l] > 0.5f);
    if (hsel) { sel_pos[b * L + cum] = l; ++cum; }
    chunk_ids[b * L + l] = cum - 1;
  }
  for (int s = cum; s < L; ++s) sel_pos[b * L + s] = -1;
}

// z[b,s,:D] = x[b,pos_s,:] (or 0), z[b,s,D:] = pad
__global__ void build_z(const float* __restrict__ x, const float* __restrict__ pad,
                        const int* __restrict__ sel_pos, float* __restrict__ z,
                        int L, int D, int PD, size_t tot)
{
  size_t idx = (size_t)blockIdx.x * 256 + threadIdx.x;
  if (idx >= tot) return;
  int j = (int)(idx % PD); size_t t = idx / PD;
  int s = (int)(t % L); int b = (int)(t / L);
  float v;
  if (j < D) {
    int p = sel_pos[b * L + s];
    v = (p >= 0) ? x[(size_t)(b * L + p) * D + j] : 0.0f;
  } else {
    v = pad[j - D];
  }
  z[idx] = v;
}

// y[b,l,j] = zn[b, chunk_ids[b,l], j] + x[b,l,j]
__global__ void decompress_add(const float* __restrict__ zn, const float* __restrict__ x,
                               const int* __restrict__ chunk_ids, float* __restrict__ y,
                               int L, int D, int PD, size_t tot)
{
  size_t idx = (size_t)blockIdx.x * 256 + threadIdx.x;
  if (idx >= tot) return;
  int j = (int)(idx % D); size_t t = idx / D;
  int l = (int)(t % L); int b = (int)(t / L);
  int cid = chunk_ids[b * L + l];
  y[idx] = zn[(size_t)(b * L + cid) * PD + j] + x[idx];
}

// rope cache for processor stage (head_dim = hd)
__global__ void rope_cache_kernel(float* __restrict__ pcos, float* __restrict__ psin,
                                  int half, int hd)
{
  int l = blockIdx.x;
  int i = threadIdx.x; if (i >= half) return;
  float inv = powf(10000.0f, -(2.0f * (float)i) / (float)hd);
  float ang = (float)l * inv;
  pcos[(size_t)l * half + i] = cosf(ang);
  psin[(size_t)l * half + i] = sinf(ang);
}

// ---------------------------------------------------------------------------
// Host-side orchestration
// ---------------------------------------------------------------------------
struct Scratch {
  _Float16 *h16, *wT, *qh, *kh, *vT, *Pm, *om, *t16;
  float *qkv, *S, *O, *g, *u;
};

static void run_block(hipStream_t st, float* x, int D, int I, int hd,
                      const float* an, const float* wqkv, const float* wo,
                      const float* fn, const float* w1, const float* w3, const float* w2,
                      const float* cb, const float* sb, const Scratch& w)
{
  const int B = 2, L = 512, H = 12, BL = 1024;
  const int D3 = 3 * D;
  // h = rmsnorm(x, an)
  rmsnorm_kernel<<<BL, 256, 0, st>>>(x, an, nullptr, w.h16, D);
  // qkv = h @ wqkv
  { size_t n = (size_t)D * D3;
    transpose_f32_to_f16<<<dim3((unsigned)((n + 255) / 256)), 256, 0, st>>>(wqkv, w.wT, D, D3); }
  gemm_f16_wmma<<<dim3((D3 + 63) / 64, BL / 32, 1), 32, 0, st>>>(
      w.h16, w.wT, w.qkv, BL, D3, D, 0, 0, 0, nullptr, 0, 0);
  // split heads + rope, v transposed
  rope_split_kernel<<<dim3(L, H, B), hd, 0, st>>>(w.qkv, cb, sb, w.qh, w.kh, w.vT, D, hd, H, L);
  // scores = q @ k^T  (batched over B*H)
  gemm_f16_wmma<<<dim3((L + 63) / 64, L / 32, B * H), 32, 0, st>>>(
      w.qh, w.kh, w.S, L, L, hd,
      (long long)L * hd, (long long)L * hd, (long long)L * L, nullptr, 0, 0);
  softmax_causal<<<dim3(L, B * H), 256, 0, st>>>(w.S, w.Pm, L, 1.0f / sqrtf((float)hd));
  // O = P @ V
  gemm_f16_wmma<<<dim3((hd + 63) / 64, L / 32, B * H), 32, 0, st>>>(
      w.Pm, w.vT, w.O, L, hd, L,
      (long long)L * L, (long long)hd * L, (long long)L * hd, nullptr, 0, 0);
  { size_t n = (size_t)B * H * L * hd;
    merge_heads<<<dim3((unsigned)((n + 255) / 256)), 256, 0, st>>>(w.O, w.om, H, L, hd, D, n); }
  // x += O @ wo
  { size_t n = (size_t)D * D;
    transpose_f32_to_f16<<<dim3((unsigned)((n + 255) / 256)), 256, 0, st>>>(wo, w.wT, D, D); }
  gemm_f16_wmma<<<dim3((D + 63) / 64, BL / 32, 1), 32, 0, st>>>(
      w.om, w.wT, x, BL, D, D, 0, 0, 0, nullptr, 0, 1);
  // FFN
  rmsnorm_kernel<<<BL, 256, 0, st>>>(x, fn, nullptr, w.h16, D);
  { size_t n = (size_t)D * I;
    transpose_f32_to_f16<<<dim3((unsigned)((n + 255) / 256)), 256, 0, st>>>(w1, w.wT, D, I); }
  gemm_f16_wmma<<<dim3((I + 63) / 64, BL / 32, 1), 32, 0, st>>>(
      w.h16, w.wT, w.g, BL, I, D, 0, 0, 0, nullptr, 0, 0);
  { size_t n = (size_t)D * I;
    transpose_f32_to_f16<<<dim3((unsigned)((n + 255) / 256)), 256, 0, st>>>(w3, w.wT, D, I); }
  gemm_f16_wmma<<<dim3((I + 63) / 64, BL / 32, 1), 32, 0, st>>>(
      w.h16, w.wT, w.u, BL, I, D, 0, 0, 0, nullptr, 0, 0);
  { size_t n = (size_t)BL * I;
    silu_mul<<<dim3((unsigned)((n + 255) / 256)), 256, 0, st>>>(w.g, w.u, w.t16, n); }
  { size_t n = (size_t)I * D;
    transpose_f32_to_f16<<<dim3((unsigned)((n + 255) / 256)), 256, 0, st>>>(w2, w.wT, I, D); }
  gemm_f16_wmma<<<dim3((D + 63) / 64, BL / 32, 1), 32, 0, st>>>(
      w.t16, w.wT, x, BL, D, I, 0, 0, 0, nullptr, 0, 1);
}

extern "C" void kernel_launch(void* const* d_in, const int* in_sizes, int n_in,
                              void* d_out, int out_size, void* d_ws, size_t ws_size,
                              hipStream_t stream)
{
  (void)in_sizes; (void)n_in; (void)out_size; (void)ws_size;
  const int B = 2, L = 512, D = 768, I = 2048, PD = 1152, PI = 3072, H = 12;
  const int NC = 3, NP = 6, ND = 3;
  const int BL = B * L;
  const int hdC = D / H;    // 64
  const int hdP = PD / H;   // 96

  int a = 0;
  const float* x_in     = (const float*)d_in[a++];
  const float* cosb     = (const float*)d_in[a++];
  const float* sinb     = (const float*)d_in[a++];
  const float* comp_an   = (const float*)d_in[a++];
  const float* comp_wqkv = (const float*)d_in[a++];
  const float* comp_wo   = (const float*)d_in[a++];
  const float* comp_fn   = (const float*)d_in[a++];
  const float* comp_w1   = (const float*)d_in[a++];
  const float* comp_w3   = (const float*)d_in[a++];
  const float* comp_w2   = (const float*)d_in[a++];
  const float* comp_norm = (const float*)d_in[a++];
  const float* router_w1 = (const float*)d_in[a++];
  const float* router_b1 = (const float*)d_in[a++];
  const float* router_w2 = (const float*)d_in[a++];
  const float* router_b2 = (const float*)d_in[a++];
  const float* pad_dim   = (const float*)d_in[a++];
  const float* proc_an   = (const float*)d_in[a++];
  const float* proc_wqkv = (const float*)d_in[a++];
  const float* proc_wo   = (const float*)d_in[a++];
  const float* proc_fn   = (const float*)d_in[a++];
  const float* proc_w1   = (const float*)d_in[a++];
  const float* proc_w3   = (const float*)d_in[a++];
  const float* proc_w2   = (const float*)d_in[a++];
  const float* proc_norm = (const float*)d_in[a++];
  const float* dec_an    = (const float*)d_in[a++];
  const float* dec_wqkv  = (const float*)d_in[a++];
  const float* dec_wo    = (const float*)d_in[a++];
  const float* dec_fn    = (const float*)d_in[a++];
  const float* dec_w1    = (const float*)d_in[a++];
  const float* dec_w3    = (const float*)d_in[a++];
  const float* dec_w2    = (const float*)d_in[a++];
  const float* dec_norm  = (const float*)d_in[a++];

  // bump allocator over workspace
  char* wp = (char*)d_ws;
  auto alloc = [&](size_t bytes) -> void* {
    void* r = (void*)wp;
    wp += (bytes + 255) & ~(size_t)255;
    return r;
  };

  Scratch s;
  s.h16 = (_Float16*)alloc((size_t)BL * PD * sizeof(_Float16));
  s.wT  = (_Float16*)alloc((size_t)PD * (3 * PD) * sizeof(_Float16)); // max weight
  s.qkv = (float*)   alloc((size_t)BL * (3 * PD) * sizeof(float));
  size_t bhld = (size_t)B * H * L * hdP;
  s.qh  = (_Float16*)alloc(bhld * sizeof(_Float16));
  s.kh  = (_Float16*)alloc(bhld * sizeof(_Float16));
  s.vT  = (_Float16*)alloc(bhld * sizeof(_Float16));
  s.S   = (float*)   alloc((size_t)B * H * L * L * sizeof(float));
  s.Pm  = (_Float16*)alloc((size_t)B * H * L * L * sizeof(_Float16));
  s.O   = (float*)   alloc(bhld * sizeof(float));
  s.om  = (_Float16*)alloc((size_t)BL * PD * sizeof(_Float16));
  s.g   = (float*)   alloc((size_t)BL * PI * sizeof(float));
  s.u   = (float*)   alloc((size_t)BL * PI * sizeof(float));
  s.t16 = (_Float16*)alloc((size_t)BL * PI * sizeof(_Float16));
  float* bufX  = (float*)alloc((size_t)BL * D * sizeof(float));
  float* bufZ  = (float*)alloc((size_t)BL * PD * sizeof(float));
  float* bufY  = (float*)alloc((size_t)BL * D * sizeof(float));
  float* r1    = (float*)alloc((size_t)BL * D * sizeof(float));
  float* probs = (float*)alloc((size_t)BL * sizeof(float));
  int*   chunk = (int*)  alloc((size_t)BL * sizeof(int));
  int*   selp  = (int*)  alloc((size_t)BL * sizeof(int));
  float* pcos  = (float*)alloc((size_t)L * (hdP / 2) * sizeof(float));
  float* psin  = (float*)alloc((size_t)L * (hdP / 2) * sizeof(float));

  // x -> bufX (inputs must not be mutated)
  hipMemcpyAsync(bufX, x_in, (size_t)BL * D * sizeof(float),
                 hipMemcpyDeviceToDevice, stream);

  // ---- compressor blocks ----
  for (int i = 0; i < NC; ++i) {
    run_block(stream, bufX, D, I, hdC,
              comp_an + (size_t)i * D,
              comp_wqkv + (size_t)i * D * 3 * D,
              comp_wo + (size_t)i * D * D,
              comp_fn + (size_t)i * D,
              comp_w1 + (size_t)i * D * I,
              comp_w3 + (size_t)i * D * I,
              comp_w2 + (size_t)i * I * D,
              cosb, sinb, s);
  }
  // x = rmsnorm(x, comp_norm)  (in place, also f16 copy for router GEMM)
  rmsnorm_kernel<<<BL, 256, 0, stream>>>(bufX, comp_norm, bufX, s.h16, D);

  // ---- router ----
  { size_t n = (size_t)D * D;
    transpose_f32_to_f16<<<dim3((unsigned)((n + 255) / 256)), 256, 0, stream>>>(router_w1, s.wT, D, D); }
  gemm_f16_wmma<<<dim3((D + 63) / 64, BL / 32, 1), 32, 0, stream>>>(
      s.h16, s.wT, r1, BL, D, D, 0, 0, 0, router_b1, /*act=silu*/1, 0);
  router_logit<<<BL, 256, 0, stream>>>(r1, router_w2, router_b2, probs, D);
  router_scan<<<B, 1, 0, stream>>>(probs, chunk, selp, L);

  // ---- compress + pad to PD ----
  { size_t n = (size_t)B * L * PD;
    build_z<<<dim3((unsigned)((n + 255) / 256)), 256, 0, stream>>>(
        bufX, pad_dim, selp, bufZ, L, D, PD, n); }

  // ---- processor blocks (head_dim 96 rope cache) ----
  rope_cache_kernel<<<L, 64, 0, stream>>>(pcos, psin, hdP / 2, hdP);
  for (int i = 0; i < NP; ++i) {
    run_block(stream, bufZ, PD, PI, hdP,
              proc_an + (size_t)i * PD,
              proc_wqkv + (size_t)i * PD * 3 * PD,
              proc_wo + (size_t)i * PD * PD,
              proc_fn + (size_t)i * PD,
              proc_w1 + (size_t)i * PD * PI,
              proc_w3 + (size_t)i * PD * PI,
              proc_w2 + (size_t)i * PI * PD,
              pcos, psin, s);
  }
  rmsnorm_kernel<<<BL, 256, 0, stream>>>(bufZ, proc_norm, bufZ, nullptr, PD);

  // ---- decompress: y = gather(xp, chunk_ids) + x ----
  { size_t n = (size_t)B * L * D;
    decompress_add<<<dim3((unsigned)((n + 255) / 256)), 256, 0, stream>>>(
        bufZ, bufX, chunk, bufY, L, D, PD, n); }

  // ---- decoder blocks ----
  for (int i = 0; i < ND; ++i) {
    run_block(stream, bufY, D, I, hdC,
              dec_an + (size_t)i * D,
              dec_wqkv + (size_t)i * D * 3 * D,
              dec_wo + (size_t)i * D * D,
              dec_fn + (size_t)i * D,
              dec_w1 + (size_t)i * D * I,
              dec_w3 + (size_t)i * D * I,
              dec_w2 + (size_t)i * I * D,
              cosb, sinb, s);
  }
  // final rmsnorm -> output
  rmsnorm_kernel<<<BL, 256, 0, stream>>>(bufY, dec_norm, (float*)d_out, nullptr, D);
}
